// BahdanauAttention_2430951490080
// MI455X (gfx1250) — matrix operand order repaired and checked by reading, not compiled, not run on hardware
//
#include <hip/hip_runtime.h>
#include <hip/hip_bf16.h>
#include <math.h>

// Problem constants (fixed by the reference)
#define B_  32
#define T_  2048
#define H_  1024
#define BT_ (B_*T_)

typedef __attribute__((ext_vector_type(16))) __bf16 v16bf;
typedef __attribute__((ext_vector_type(8)))  __bf16 v8bf;
typedef __attribute__((ext_vector_type(8)))  float  v8f;

union AFrag { v16bf v; v8bf h[2]; };

static __device__ __forceinline__ unsigned f32_to_bf16_rne(float x) {
    unsigned u = __float_as_uint(x);
    u += 0x7fffu + ((u >> 16) & 1u);          // round-to-nearest-even
    return u >> 16;
}

// ---------------------------------------------------------------------------
// Kernel 0a: qpb[b,o] = dot(query[b,:], Wq[o,:]) + bq[o] + bv[o]
// (bv folded in: the score epilogue needs qp + bv as one constant)
// ---------------------------------------------------------------------------
__global__ __launch_bounds__(256) void qproj_kernel(
    const float* __restrict__ query, const float* __restrict__ Wq,
    const float* __restrict__ bq, const float* __restrict__ bv,
    float* __restrict__ qpb)
{
    int idx = blockIdx.x * 256 + threadIdx.x;   // B_*H_ threads
    int b = idx >> 10, o = idx & (H_ - 1);
    const float4* q4 = (const float4*)(query + (size_t)b * H_);
    const float4* w4 = (const float4*)(Wq    + (size_t)o * H_);
    float acc = 0.f;
    #pragma unroll 4
    for (int h = 0; h < H_/4; ++h) {
        float4 a = q4[h], c = w4[h];
        acc += a.x*c.x + a.y*c.y + a.z*c.z + a.w*c.w;
    }
    qpb[idx] = acc + bq[o] + bv[o];
}

// ---------------------------------------------------------------------------
// Kernel 0b: Wv (f32) -> bf16, halves weight-streaming traffic for the GEMM
// ---------------------------------------------------------------------------
__global__ __launch_bounds__(256) void cvt_wv_kernel(
    const float* __restrict__ Wv, unsigned short* __restrict__ wvb)
{
    int i = blockIdx.x * 256 + threadIdx.x;     // H_*H_ threads
    wvb[i] = (unsigned short)f32_to_bf16_rne(Wv[i]);
}

// ---------------------------------------------------------------------------
// Kernel 1: fused  score[b,t] = Vw . tanh(qpb[b] + Wv@values[b,t]) + Vb
//
// Blocking: 64 (b,t)-rows per block (4 row-tiles), staged once as bf16 in
// 129 KB dynamic LDS. 8 waves sweep the 64 o-tiles in 4 passes of 2 o-tiles
// per wave. Per k-step: 2 global B-loads (reused across 4 row-tiles from
// registers) + 4 LDS A-loads feed 8 v_wmma_f32_16x16x32_bf16.
// Wv (bf16, 2 MB) is read exactly once per block -> 2 GB total L2 traffic.
// k-loop kept rolled (#pragma unroll 1): full unroll previously spilled.
// ---------------------------------------------------------------------------
__global__ __launch_bounds__(256, 1) void score_kernel(
    const float* __restrict__ values, const unsigned short* __restrict__ wvb,
    const float* __restrict__ qpb, const float* __restrict__ Vw,
    const float* __restrict__ Vb, float* __restrict__ scores)
{
    constexpr int PAD  = 8;          // 16B row pad -> conflict-free ds_load_b128
    constexpr int LROW = H_ + PAD;   // 1032 bf16 per LDS row
    constexpr int ROWS = 64;         // rows per block
    extern __shared__ __align__(16) unsigned short dsmem[];   // ROWS*LROW bf16
    __shared__ float sred[8][4][16];

    const int rbase = blockIdx.x * ROWS;   // 64 | T, so whole block is one batch b
    const int b     = rbase / T_;
    const int tid   = threadIdx.x;

    // ---- stage A: 64 rows x 1024 f32 -> bf16 pairs in LDS (coalesced b64) ----
    const float2* src2 = (const float2*)(values + (size_t)rbase * H_);
    for (int i = tid; i < ROWS * (H_/2); i += 256) {
        int r = i >> 9, k2 = (i & 511) << 1;
        float2 f = src2[i];
        unsigned lo = f32_to_bf16_rne(f.x), hi = f32_to_bf16_rne(f.y);
        *(unsigned*)(dsmem + (size_t)r * LROW + k2) = lo | (hi << 16);
    }
    __syncthreads();

    const int w      = tid >> 5;     // wave 0..7
    const int l      = tid & 31;     // lane
    const int nrow   = l & 15;       // M-row (A) / N-col (B) owned by this lane
    const int hiHalf = l >> 4;       // 0: lanes 0-15, 1: lanes 16-31

    // A-fragment base: ISA 16-bit A 16x32 layout — lane<16 holds K {0..7,16..23},
    // lane>=16 holds K {8..15,24..31}, as two contiguous 8-element groups.
    const unsigned short* aBase = dsmem + nrow * LROW + hiHalf * 8;

    float rowacc[4][8];
    #pragma unroll
    for (int rt = 0; rt < 4; ++rt)
        #pragma unroll
        for (int v = 0; v < 8; ++v) rowacc[rt][v] = 0.f;

    const v8f zero = {0.f,0.f,0.f,0.f,0.f,0.f,0.f,0.f};

    #pragma unroll 1
    for (int p = 0; p < 4; ++p) {            // 4 passes over the o dimension
        const int ob0 = (p * 16 + w * 2) * 16;   // wave's first o this pass
        __builtin_prefetch(wvb + (size_t)(ob0 + nrow) * H_, 0, 1);

        v8f acc[2][4];
        #pragma unroll
        for (int ot = 0; ot < 2; ++ot)
            #pragma unroll
            for (int rt = 0; rt < 4; ++rt) acc[ot][rt] = zero;

        // keep rolled: fully unrolling 32 iterations spilled A/B prefetches
        #pragma unroll 1
        for (int ks = 0; ks < H_; ks += 32) {
            // load both B-fragments first (live across the rt loop)
            AFrag bb[2];
            #pragma unroll
            for (int ot = 0; ot < 2; ++ot) {
                const unsigned short* bp =
                    wvb + (size_t)(ob0 + ot * 16 + nrow) * H_ + ks + hiHalf * 8;
                bb[ot].h[0] = *(const v8bf*)(bp);
                bb[ot].h[1] = *(const v8bf*)(bp + 16);
            }
            // per row-tile: one A-fragment, immediately consumed by 2 WMMAs
            #pragma unroll
            for (int rt = 0; rt < 4; ++rt) {
                const unsigned short* ap = aBase + rt * 16 * LROW + ks;
                AFrag a;
                a.h[0] = *(const v8bf*)(ap);
                a.h[1] = *(const v8bf*)(ap + 16);
                acc[0][rt] = __builtin_amdgcn_wmma_f32_16x16x32_bf16(
                    false, a.v, false, bb[0].v, (short)0, acc[0][rt], false, false);
                acc[1][rt] = __builtin_amdgcn_wmma_f32_16x16x32_bf16(
                    false, a.v, false, bb[1].v, (short)0, acc[1][rt], false, false);
            }
        }

        // epilogue of this pass: tanh, scale by Vw, fold into row partials.
        // C/D layout: lane l element v -> row M = v + 8*hiHalf, col N = nrow
        #pragma unroll
        for (int ot = 0; ot < 2; ++ot) {
            int o = ob0 + ot * 16 + nrow;
            float qv = qpb[b * H_ + o];
            float vw = Vw[o];
            #pragma unroll
            for (int rt = 0; rt < 4; ++rt)
                #pragma unroll
                for (int v = 0; v < 8; ++v)
                    rowacc[rt][v] += tanhf(acc[ot][rt][v] + qv) * vw;
        }
    }

    // reduce across the 16 lanes of each half (masks <16 stay inside the half)
    #pragma unroll
    for (int rt = 0; rt < 4; ++rt)
        #pragma unroll
        for (int v = 0; v < 8; ++v) {
            #pragma unroll
            for (int m = 1; m < 16; m <<= 1)
                rowacc[rt][v] += __shfl_xor(rowacc[rt][v], m, 32);
        }
    if (nrow == 0) {
        #pragma unroll
        for (int rt = 0; rt < 4; ++rt)
            #pragma unroll
            for (int v = 0; v < 8; ++v)
                sred[w][rt][v + 8 * hiHalf] = rowacc[rt][v];
    }
    __syncthreads();

    if (tid < 64) {              // cross-wave o-reduction, add Vb, emit scores
        int rt = tid >> 4, m = tid & 15;
        float s = Vb[0];
        #pragma unroll
        for (int ww = 0; ww < 8; ++ww) s += sred[ww][rt][m];
        scores[rbase + rt * 16 + m] = s;
    }
}

// ---------------------------------------------------------------------------
// Kernel 2: softmax over T per batch; writes attention_weights into d_out
// ---------------------------------------------------------------------------
__global__ __launch_bounds__(256) void softmax_kernel(
    const float* __restrict__ scores, float* __restrict__ attn)
{
    __shared__ float red[256];
    const int b = blockIdx.x, tid = threadIdx.x;
    const float* s = scores + (size_t)b * T_;
    float x[8];
    float mx = -1e30f;
    #pragma unroll
    for (int i = 0; i < 8; ++i) { x[i] = s[tid + i * 256]; mx = fmaxf(mx, x[i]); }
    red[tid] = mx; __syncthreads();
    for (int st = 128; st > 0; st >>= 1) {
        if (tid < st) red[tid] = fmaxf(red[tid], red[tid + st]);
        __syncthreads();
    }
    mx = red[0]; __syncthreads();
    float sum = 0.f;
    #pragma unroll
    for (int i = 0; i < 8; ++i) { x[i] = __expf(x[i] - mx); sum += x[i]; }
    red[tid] = sum; __syncthreads();
    for (int st = 128; st > 0; st >>= 1) {
        if (tid < st) red[tid] += red[tid + st];
        __syncthreads();
    }
    float inv = 1.f / red[0];
    #pragma unroll
    for (int i = 0; i < 8; ++i) attn[(size_t)b * T_ + tid + i * 256] = x[i] * inv;
}

// ---------------------------------------------------------------------------
// Kernel 3: partial context over a T-chunk (deterministic: no float atomics)
// grid = (B, 8); 256 threads x float4 covers all H=1024
// ---------------------------------------------------------------------------
__global__ __launch_bounds__(256) void ctx_partial_kernel(
    const float* __restrict__ values, const float* __restrict__ attn,
    float* __restrict__ partial)
{
    const int b = blockIdx.x, c = blockIdx.y;
    const int h4 = threadIdx.x * 4;
    const int TC = T_ / 8;
    float4 acc = {0.f, 0.f, 0.f, 0.f};
    const float* vp = values + ((size_t)b * T_ + (size_t)c * TC) * H_ + h4;
    const float* ap = attn + (size_t)b * T_ + c * TC;
    for (int t = 0; t < TC; ++t) {
        float wgt = ap[t];
        float4 v = *(const float4*)(vp + (size_t)t * H_);
        acc.x += wgt * v.x; acc.y += wgt * v.y;
        acc.z += wgt * v.z; acc.w += wgt * v.w;
    }
    *(float4*)(partial + ((size_t)c * B_ + b) * H_ + h4) = acc;
}

// ---------------------------------------------------------------------------
// Kernel 4: reduce the 8 partials -> context_vector in d_out
// ---------------------------------------------------------------------------
__global__ __launch_bounds__(256) void ctx_reduce_kernel(
    const float* __restrict__ partial, float* __restrict__ ctx)
{
    int i = blockIdx.x * 256 + threadIdx.x;   // B_*H_/4 threads
    size_t idx = (size_t)i * 4;
    float4 acc = *(const float4*)(partial + idx);
    #pragma unroll
    for (int c = 1; c < 8; ++c) {
        float4 p = *(const float4*)(partial + (size_t)c * B_ * H_ + idx);
        acc.x += p.x; acc.y += p.y; acc.z += p.z; acc.w += p.w;
    }
    *(float4*)(ctx + idx) = acc;
}

// ---------------------------------------------------------------------------
extern "C" void kernel_launch(void* const* d_in, const int* in_sizes, int n_in,
                              void* d_out, int out_size, void* d_ws, size_t ws_size,
                              hipStream_t stream)
{
    const float* query = (const float*)d_in[0];
    const float* values= (const float*)d_in[1];
    const float* Wq    = (const float*)d_in[2];
    const float* bq    = (const float*)d_in[3];
    const float* Wv    = (const float*)d_in[4];
    const float* bv    = (const float*)d_in[5];
    const float* Vw    = (const float*)d_in[6];
    const float* Vb    = (const float*)d_in[7];

    char* ws = (char*)d_ws;
    float*          qpb     = (float*)(ws);                 // B*H f32      = 128 KB
    float*          scores  = (float*)(ws + 131072);        // B*T f32      = 256 KB
    float*          partial = (float*)(ws + 393216);        // 8*B*H f32    = 1 MB
    unsigned short* wvb     = (unsigned short*)(ws + 1441792); // H*H bf16  = 2 MB

    float* ctx  = (float*)d_out;            // context_vector [B,H]
    float* attn = (float*)d_out + B_ * H_;  // attention_weights [B,T,1]

    const size_t score_lds = (size_t)64 * (H_ + 8) * sizeof(unsigned short); // 129 KB

    qproj_kernel      <<<B_ * H_ / 256, 256, 0, stream>>>(query, Wq, bq, bv, qpb);
    cvt_wv_kernel     <<<H_ * H_ / 256, 256, 0, stream>>>(Wv, wvb);
    score_kernel      <<<BT_ / 64, 256, score_lds, stream>>>(values, wvb, qpb, Vw, Vb, scores);
    softmax_kernel    <<<B_,            256, 0, stream>>>(scores, attn);
    ctx_partial_kernel<<<dim3(B_, 8),   256, 0, stream>>>(values, attn, partial);
    ctx_reduce_kernel <<<(B_ * H_ / 4) / 256, 256, 0, stream>>>(partial, ctx);
}
